// LSTM2FC_83253646066033
// MI455X (gfx1250) — matrix operand order repaired and checked by reading, not compiled, not run on hardware
//
#include <hip/hip_runtime.h>
#include <math.h>

// LSTM2FC fused kernel for gfx1250 (MI455X), v2.
// One workgroup = 16 batch rows (WMMA M), loops all T=256 timesteps.
// 8 waves (wave32); wave w owns 3 WMMA N-tiles (48 gate columns).
// All weight B-fragments live in VGPRs for the whole recurrence (33 x v16h = 264 VGPRs/wave);
// only activations (A-fragments), gates, and cell state touch LDS (~47 KB/WG).
// GEMMs: v_wmma_f32_16x16x32_f16. Nonlinearities: hardware v_tanh_f32.

typedef __attribute__((ext_vector_type(16))) _Float16 v16h;
typedef __attribute__((ext_vector_type(8)))  float    v8f;

#define TT      256
#define DD      64
#define HH      96
#define G4      384     // 4*H
#define ROWS    16      // batch rows per workgroup (WMMA M)
#define THREADS 256     // 8 waves; wave w -> n-tiles 3w..3w+2

__device__ __forceinline__ v16h ldv16(const _Float16* p) {
  return *(const v16h*)p;   // 32B contiguous -> 2x ds_load_b128
}

__device__ __forceinline__ v8f wmma_f16(v16h a, v16h b, v8f c) {
  // D = A(16x32 f16) * B(32x16 f16) + C(16x16 f32)
  return __builtin_amdgcn_wmma_f32_16x16x32_f16(false, a, false, b, (short)0, c, false, false);
}

// A-fragment (16x32 f16) LDS offset for matrix element (row, k):
//  lane<16: row=lane, halves 0-7 -> K 0-7,  halves 8-15 -> K 16-23
//  lane>=16: row=lane-16, halves 0-7 -> K 8-15, halves 8-15 -> K 24-31
__device__ __forceinline__ int a_off(int row, int kk) {
  int kt   = kk >> 5;
  int r    = kk & 31;
  int lane = (row & 15) + (((r >> 3) & 1) << 4);
  int h    = (r & 7) + ((r & 16) >> 1);
  return kt * 512 + lane * 16 + h;
}

// Hardware transcendental tanh (CDNA5 trans op). Trailing v_nop covers the
// TRANS->use co-execution hazard the compiler cannot see inside inline asm.
__device__ __forceinline__ float tanh_hw(float x) {
  float r;
  asm volatile("v_tanh_f32 %0, %1\n\tv_nop" : "=v"(r) : "v"(x));
  return r;
}
__device__ __forceinline__ float sigmoid_hw(float x) {
  return 0.5f * tanh_hw(0.5f * x) + 0.5f;
}

// Load one WMMA B-fragment for this lane directly from global weight matrix
// W[G4 x Kdim] (row-major; gates = act @ W^T, so B[k][n] = W[n][k]).
// Lane's 16 halves: lanes 0-15: N=lane&15, K=kt*32+{0..7,16..23};
// lanes 16-31: K=kt*32+{8..15,24..31}.
__device__ __forceinline__ v16h loadB(const float* __restrict__ W, int Kdim,
                                      int ntile, int kt, int lane) {
  const int n = ntile * 16 + (lane & 15);
  const int k = kt * 32 + ((lane & 16) ? 8 : 0);
  const float* p = &W[n * Kdim + k];
  v16h r;
#pragma unroll
  for (int h = 0; h < 16; ++h) r[h] = (_Float16)p[h + ((h >= 8) ? 8 : 0)];
  return r;
}

__global__ __launch_bounds__(THREADS, 1)
void lstm2fc_kernel(const float* __restrict__ x,
                    const float* __restrict__ Wih0, const float* __restrict__ Whh0,
                    const float* __restrict__ b0,
                    const float* __restrict__ Wih1, const float* __restrict__ Whh1,
                    const float* __restrict__ b1,
                    const float* __restrict__ fcW1, const float* __restrict__ fcb1,
                    const float* __restrict__ fcW2, const float* __restrict__ fcb2,
                    const float* __restrict__ fcW3, const float* __restrict__ fcb3,
                    float* __restrict__ out) {
  // ---- LDS: activations + state only (~47 KB) ----
  __shared__ __attribute__((aligned(32))) _Float16 sAx [2 * 512];  // x_t A-frags
  __shared__ __attribute__((aligned(32))) _Float16 sAh1[3 * 512];  // h1 A-frags
  __shared__ __attribute__((aligned(32))) _Float16 sAh2[3 * 512];  // h2 A-frags
  __shared__ float sGates[G4 * ROWS];  // col-major [col][row]; reused for final h2 / MLP
  __shared__ float sC1[HH * ROWS];     // c1 state, col-major
  __shared__ float sC2[HH * ROWS];     // c2 state
  __shared__ float sB0[G4];
  __shared__ float sB1[G4];

  const int tid  = threadIdx.x;
  const int lane = tid & 31;
  const int nt0  = (tid >> 5) * 3;     // first n-tile of this wave
  const size_t brow = (size_t)blockIdx.x * ROWS;

  // ---- one-time: weights straight from global into VGPRs (B-fragment layout) ----
  v16h rWih0[3][2], rWhh0[3][3], rWih1[3][3], rWhh1[3][3];
#pragma unroll
  for (int j = 0; j < 3; ++j) {
    const int nt = nt0 + j;
#pragma unroll
    for (int kt = 0; kt < 2; ++kt) rWih0[j][kt] = loadB(Wih0, DD, nt, kt, lane);
#pragma unroll
    for (int kt = 0; kt < 3; ++kt) {
      rWhh0[j][kt] = loadB(Whh0, HH, nt, kt, lane);
      rWih1[j][kt] = loadB(Wih1, HH, nt, kt, lane);
      rWhh1[j][kt] = loadB(Whh1, HH, nt, kt, lane);
    }
  }

  // ---- init LDS state ----
  for (int e = tid; e < G4; e += THREADS) { sB0[e] = b0[e]; sB1[e] = b1[e]; }
  for (int e = tid; e < HH * ROWS; e += THREADS) { sC1[e] = 0.f; sC2[e] = 0.f; }
  for (int e = tid; e < 3 * 512; e += THREADS) {
    sAh1[e] = (_Float16)0.f; sAh2[e] = (_Float16)0.f;
  }
  // stage x_0
  for (int e = tid; e < ROWS * DD; e += THREADS) {
    int row = e >> 6, kk = e & 63;
    sAx[a_off(row, kk)] = (_Float16)x[((brow + row) * TT + 0) * DD + kk];
  }
  __syncthreads();

  // ---- recurrence over T (4 barriers per step) ----
  for (int t = 0; t < TT; ++t) {
    // ---- layer 0 GEMM: gates = x_t @ Wih0^T + h1 @ Whh0^T ----
    {
      v16h a0 = ldv16(&sAx [         lane * 16]);
      v16h a1 = ldv16(&sAx [ 512 +   lane * 16]);
      v16h h0 = ldv16(&sAh1[         lane * 16]);
      v16h h1 = ldv16(&sAh1[ 512 +   lane * 16]);
      v16h h2 = ldv16(&sAh1[1024 +   lane * 16]);
#pragma unroll
      for (int j = 0; j < 3; ++j) {
        const int nt = nt0 + j;
        v8f acc = {};
        acc = wmma_f16(a0, rWih0[j][0], acc);
        acc = wmma_f16(a1, rWih0[j][1], acc);
        acc = wmma_f16(h0, rWhh0[j][0], acc);
        acc = wmma_f16(h1, rWhh0[j][1], acc);
        acc = wmma_f16(h2, rWhh0[j][2], acc);
        // C layout: VGPR r, lane L -> m = r + 8*(L>>4), n = L&15 => contiguous col-major store
        float* p = &sGates[(nt * 16 + (lane & 15)) * 16 + ((lane >> 4) << 3)];
#pragma unroll
        for (int r = 0; r < 8; ++r) p[r] = acc[r];
      }
    }
    __syncthreads();

    // ---- layer 0 elementwise: LSTM cell, writes h1 into A-layout ----
    for (int e = tid; e < ROWS * HH; e += THREADS) {
      int col = e >> 4, row = e & 15;
      float gi = sGates[(col         ) * 16 + row] + sB0[col];
      float gf = sGates[(col +     HH) * 16 + row] + sB0[col +     HH];
      float gg = sGates[(col + 2 * HH) * 16 + row] + sB0[col + 2 * HH];
      float go = sGates[(col + 3 * HH) * 16 + row] + sB0[col + 3 * HH];
      float iv = sigmoid_hw(gi), fv = sigmoid_hw(gf);
      float gv = tanh_hw(gg),    ov = sigmoid_hw(go);
      float c  = fv * sC1[e] + iv * gv;
      sC1[e] = c;
      sAh1[a_off(row, col)] = (_Float16)(ov * tanh_hw(c));
    }
    __syncthreads();

    // ---- layer 1 GEMM: gates = h1 @ Wih1^T + h2 @ Whh1^T ----
    {
      v16h i0 = ldv16(&sAh1[         lane * 16]);
      v16h i1 = ldv16(&sAh1[ 512 +   lane * 16]);
      v16h i2 = ldv16(&sAh1[1024 +   lane * 16]);
      v16h r0 = ldv16(&sAh2[         lane * 16]);
      v16h r1 = ldv16(&sAh2[ 512 +   lane * 16]);
      v16h r2 = ldv16(&sAh2[1024 +   lane * 16]);
#pragma unroll
      for (int j = 0; j < 3; ++j) {
        const int nt = nt0 + j;
        v8f acc = {};
        acc = wmma_f16(i0, rWih1[j][0], acc);
        acc = wmma_f16(i1, rWih1[j][1], acc);
        acc = wmma_f16(i2, rWih1[j][2], acc);
        acc = wmma_f16(r0, rWhh1[j][0], acc);
        acc = wmma_f16(r1, rWhh1[j][1], acc);
        acc = wmma_f16(r2, rWhh1[j][2], acc);
        float* p = &sGates[(nt * 16 + (lane & 15)) * 16 + ((lane >> 4) << 3)];
#pragma unroll
        for (int r = 0; r < 8; ++r) p[r] = acc[r];
      }
    }
    __syncthreads();

    // ---- layer 1 elementwise + stage x_{t+1}; on last step stash h2 for MLP ----
    for (int e = tid; e < ROWS * HH; e += THREADS) {
      int col = e >> 4, row = e & 15;
      float gi = sGates[(col         ) * 16 + row] + sB1[col];
      float gf = sGates[(col +     HH) * 16 + row] + sB1[col +     HH];
      float gg = sGates[(col + 2 * HH) * 16 + row] + sB1[col + 2 * HH];
      float go = sGates[(col + 3 * HH) * 16 + row] + sB1[col + 3 * HH];
      float iv = sigmoid_hw(gi), fv = sigmoid_hw(gf);
      float gv = tanh_hw(gg),    ov = sigmoid_hw(go);
      float c  = fv * sC2[e] + iv * gv;
      sC2[e] = c;
      float h2v = ov * tanh_hw(c);
      sAh2[a_off(row, col)] = (_Float16)h2v;
      if (t == TT - 1) sGates[e] = h2v;   // same-thread overwrite of own 'i' cell: safe
    }
    if (t + 1 < TT) {
      const int tn = t + 1;
      for (int e = tid; e < ROWS * DD; e += THREADS) {
        int row = e >> 6, kk = e & 63;
        const float* px = &x[((brow + row) * TT + tn) * DD + kk];
        sAx[a_off(row, kk)] = (_Float16)(*px);
        if (tn + 1 < TT) __builtin_prefetch(px + DD, 0, 0);
      }
    }
    __syncthreads();
  }

  // ---- MLP head on final h2 (f32 VALU; tiny) ----
  for (int e = tid; e < ROWS * HH; e += THREADS) {       // z1 -> sC1
    int col = e >> 4, row = e & 15;
    float s = fcb1[col];
    for (int k = 0; k < HH; ++k) s += sGates[k * 16 + row] * fcW1[col * HH + k];
    sC1[e] = (s > 0.f) ? s : 0.01f * s;
  }
  __syncthreads();
  for (int e = tid; e < ROWS * HH; e += THREADS) {       // z2 -> sC2
    int col = e >> 4, row = e & 15;
    float s = fcb2[col];
    for (int k = 0; k < HH; ++k) s += sC1[k * 16 + row] * fcW2[col * HH + k];
    sC2[e] = (s > 0.f) ? s : 0.01f * s;
  }
  __syncthreads();
  if (tid < ROWS) {                                      // out[row]
    float s = fcb3[0];
    for (int k = 0; k < HH; ++k) s += sC2[k * 16 + tid] * fcW3[k];
    out[brow + tid] = s;
  }
}

extern "C" void kernel_launch(void* const* d_in, const int* in_sizes, int n_in,
                              void* d_out, int out_size, void* d_ws, size_t ws_size,
                              hipStream_t stream) {
  (void)n_in; (void)out_size; (void)d_ws; (void)ws_size;
  const float* x    = (const float*)d_in[0];
  const float* Wih0 = (const float*)d_in[1];
  const float* Whh0 = (const float*)d_in[2];
  const float* b0   = (const float*)d_in[3];
  const float* Wih1 = (const float*)d_in[4];
  const float* Whh1 = (const float*)d_in[5];
  const float* b1   = (const float*)d_in[6];
  const float* fcW1 = (const float*)d_in[7];
  const float* fcb1 = (const float*)d_in[8];
  const float* fcW2 = (const float*)d_in[9];
  const float* fcb2 = (const float*)d_in[10];
  const float* fcW3 = (const float*)d_in[11];
  const float* fcb3 = (const float*)d_in[12];
  float* out = (float*)d_out;

  const int Bn = in_sizes[0] / (TT * DD);   // 4096
  dim3 grid(Bn / ROWS);                      // 256 workgroups
  dim3 block(THREADS);                       // 8 waves
  lstm2fc_kernel<<<grid, block, 0, stream>>>(x, Wih0, Whh0, b0, Wih1, Whh1, b1,
                                             fcW1, fcb1, fcW2, fcb2, fcW3, fcb3, out);
}
